// WaveNet_14791867367772
// MI455X (gfx1250) — matrix operand order repaired and checked by reading, not compiled, not run on hardware
//
#include <hip/hip_runtime.h>

// ---------------- problem constants ----------------
#define B_   4
#define T_   16000
#define TF_  80
#define NC_  80
#define NR_  64
#define NS_  256
#define NO_  256
#define NL_  16
#define KUP_ 800
#define STR_ 200

#define TILE_N 64          // time steps per block tile
#define KF     224         // fused K for layer GEMM: 64 prev + 64 cur + 96 cond(pad)
#define LACT_S 232         // lds row stride (bf16) for fused act tile (bank-friendly)
#define LGAT_S 72          // lds row stride for gated acts
#define LOUT_S 264         // lds row stride for output-head tiles

typedef unsigned short u16;
typedef __attribute__((ext_vector_type(16))) __bf16 v16bf;
typedef __attribute__((ext_vector_type(8)))  float  v8f;
typedef int v4i __attribute__((vector_size(16)));   // matches builtin's V4i pointee

// ---- optional CDNA5 async global->LDS path (guarded; falls back to VGPR staging) ----
#if defined(__has_builtin)
#  if __has_builtin(__builtin_amdgcn_global_load_async_to_lds_b128)
#    define HAVE_ASYNC_LDS 1
#  endif
#endif
#ifndef HAVE_ASYNC_LDS
#  define HAVE_ASYNC_LDS 0
#endif

#if HAVE_ASYNC_LDS
#define AS1 __attribute__((address_space(1)))
#define AS3 __attribute__((address_space(3)))
__device__ __forceinline__ void async_cp16(const void* g, void* l) {
  __builtin_amdgcn_global_load_async_to_lds_b128((AS1 v4i*)(g), (AS3 v4i*)(l), 0, 0);
}
#  if __has_builtin(__builtin_amdgcn_s_wait_asynccnt)
#    define WAIT_ASYNC() __builtin_amdgcn_s_wait_asynccnt(0)
#  else
#    define WAIT_ASYNC() asm volatile("s_wait_asynccnt 0" ::: "memory")
#  endif
#else
#  define WAIT_ASYNC() ((void)0)
#endif

union FragU { uint4 q[2]; v16bf v; };

__device__ __forceinline__ u16 f2bf(float x) {
  unsigned u = __float_as_uint(x);
  u = (u + 0x7FFFu + ((u >> 16) & 1u)) >> 16;   // round-to-nearest-even
  return (u16)u;
}
__device__ __forceinline__ float bf2f(u16 s) { return __uint_as_float(((unsigned)s) << 16); }
__device__ __forceinline__ unsigned pack2(float a, float b) {
  return (unsigned)f2bf(a) | ((unsigned)f2bf(b) << 16);
}

// A-fragment (16x32 bf16): lane row = M, 16 contiguous K values
__device__ __forceinline__ v16bf ldA(const u16* p) {
  FragU f; f.q[0] = *(const uint4*)p; f.q[1] = *(const uint4*)(p + 8); return f.v;
}
// B-fragment (32x16 bf16): lane col = N, 16 contiguous K values
__device__ __forceinline__ v16bf ldB(const u16* p) {
  FragU f; f.q[0] = *(const uint4*)p; f.q[1] = *(const uint4*)(p + 8); return f.v;
}
__device__ __forceinline__ v8f ld8f(const float* p) {
  float4 a = *(const float4*)p, b = *(const float4*)(p + 4);
  v8f r; r[0]=a.x; r[1]=a.y; r[2]=a.z; r[3]=a.w; r[4]=b.x; r[5]=b.y; r[6]=b.z; r[7]=b.w;
  return r;
}
#define WMMA_BF16(A,Bf,C) \
  __builtin_amdgcn_wmma_f32_16x16x32_bf16(false,(A),false,(Bf),(short)0,(C),false,false)

// ---------------- weight packing / transposes ----------------
__global__ void pack_kernel(
    const float* __restrict__ embed, const float* __restrict__ up_w,
    const float* __restrict__ cond_w, const float* __restrict__ cond_b,
    const float* __restrict__ dil_w, const float* __restrict__ dil_b,
    const float* __restrict__ res_w, const float* __restrict__ skip_w,
    const float* __restrict__ out_w, const float* __restrict__ end_w,
    u16* __restrict__ Wl, float* __restrict__ bl, u16* __restrict__ Wr,
    u16* __restrict__ Wk, u16* __restrict__ Wo, u16* __restrict__ We,
    u16* __restrict__ Emb, float* __restrict__ upT)
{
  const size_t gid = (size_t)blockIdx.x * blockDim.x + threadIdx.x;
  const size_t gsz = (size_t)gridDim.x * blockDim.x;

  // fused layer weights: [dil tap0 | dil tap1 | cond slice | zero pad] (128 x 224 per layer)
  for (size_t i = gid; i < (size_t)NL_ * 128 * KF; i += gsz) {
    int L = (int)(i / (128 * KF)); int r = (int)((i / KF) % 128); int c = (int)(i % KF);
    float v;
    if      (c < 64)  v = dil_w[(((size_t)L * 128 + r) * 64 + c) * 2 + 0];
    else if (c < 128) v = dil_w[(((size_t)L * 128 + r) * 64 + (c - 64)) * 2 + 1];
    else if (c < 208) v = cond_w[((size_t)L * 128 + r) * 80 + (c - 128)];
    else              v = 0.0f;
    Wl[i] = f2bf(v);
  }
  for (size_t i = gid; i < (size_t)NL_ * 128; i += gsz) bl[i] = dil_b[i] + cond_b[i];
  for (size_t i = gid; i < (size_t)(NL_ - 1) * 64 * 64; i += gsz) Wr[i] = f2bf(res_w[i]);
  for (size_t i = gid; i < (size_t)NL_ * 256 * 64; i += gsz) Wk[i] = f2bf(skip_w[i]);
  for (size_t i = gid; i < (size_t)256 * 256; i += gsz) Wo[i] = f2bf(out_w[i]);
  for (size_t i = gid; i < (size_t)256 * 256; i += gsz) We[i] = f2bf(end_w[i]);
  for (size_t i = gid; i < (size_t)256 * 64;  i += gsz) Emb[i] = f2bf(embed[i]);
  // up_w (i,o,k) -> upT (i,k,o) so consecutive o is coalesced
  for (size_t i = gid; i < (size_t)NC_ * KUP_ * NC_; i += gsz) {
    int ii = (int)(i / (KUP_ * NC_)); int k = (int)((i / NC_) % KUP_); int o = (int)(i % NC_);
    upT[i] = up_w[((size_t)ii * NC_ + o) * KUP_ + k];
  }
}

// ---------------- transposed-conv upsampling -> cond (B,T,96) bf16 ----------------
__global__ __launch_bounds__(256) void upsample_kernel(
    const float* __restrict__ feat, const float* __restrict__ upT,
    const float* __restrict__ up_b, u16* __restrict__ cond)
{
  __shared__ float f[NC_ * TF_];           // features[b] : 25.6 KB
  const int tid = threadIdx.x;
  const int b = blockIdx.y;
  const int t0 = blockIdx.x * 8;

  for (int i = tid; i < NC_ * TF_; i += 256) f[i] = feat[(size_t)b * NC_ * TF_ + i];
  __syncthreads();

  for (int u = tid; u < 8 * 96; u += 256) {
    const int n = u / 96, o = u % 96;
    const int t = t0 + n;
    float acc = 0.0f;
    if (o < NC_) {
      acc = up_b[o];
      for (int s = t / STR_; s >= 0; --s) {
        const int k = t - STR_ * s;
        if (k > KUP_ - 1) break;
        const float* wp = upT + (size_t)k * NC_ + o;   // + i*800*80 per channel
        for (int i = 0; i < NC_; ++i)
          acc = fmaf(f[i * TF_ + s], wp[(size_t)i * KUP_ * NC_], acc);
      }
    }
    cond[((size_t)b * T_ + t) * 96 + o] = f2bf(acc);
  }
}

// ---------------- embedding gather -> x0 (B,T,64) bf16 ----------------
__global__ void embed_kernel(const int* __restrict__ fwd, const u16* __restrict__ Emb,
                             u16* __restrict__ x0)
{
  const size_t gsz = (size_t)gridDim.x * blockDim.x;
  for (size_t u = (size_t)blockIdx.x * blockDim.x + threadIdx.x;
       u < (size_t)B_ * T_ * 8; u += gsz) {
    const size_t bt = u >> 3; const int seg = (int)(u & 7);
    const int idx = fwd[bt];
    *(uint4*)&x0[bt * 64 + seg * 8] = *(const uint4*)&Emb[(size_t)idx * 64 + seg * 8];
  }
}

// ---------------- one WaveNet layer (fused dilated+cond GEMM, gate, res, skip) ----------------
__global__ __launch_bounds__(128) void layer_kernel(
    const u16* __restrict__ Wl, const float* __restrict__ bl,
    const u16* __restrict__ Wr, const float* __restrict__ rb,
    const u16* __restrict__ Wk, const float* __restrict__ kb,
    const u16* __restrict__ cond, const u16* __restrict__ xin,
    u16* __restrict__ xout, float* __restrict__ skip,
    int layer, int dil, int first, int last)
{
  __shared__ u16 lact[TILE_N * LACT_S];   // fused B tile [n][224]
  __shared__ u16 lgat[TILE_N * LGAT_S];   // gated acts   [n][64]
  const int tid = threadIdx.x;
  const int b  = blockIdx.y;
  const int t0 = blockIdx.x * TILE_N;

  // prefetch this block's skip-accumulator rows (RMW'd at kernel tail)
  if (!first) {
    for (int n = tid; n < TILE_N; n += 128)
      __builtin_prefetch(skip + ((size_t)b * T_ + t0 + n) * 256, 0, 3);
  }

  // stage: [0:64]=x(t-d), [64:128]=x(t), [128:224]=cond(t) (96 incl. zero pad)
  for (int u = tid; u < TILE_N * 28; u += 128) {
    const int n = u / 28, seg = u % 28;
    const int t = t0 + n;
    u16* dst;
    const u16* src = nullptr;
    if (seg < 8) {
      const int tp = t - dil;
      dst = &lact[n * LACT_S + seg * 8];
      if (tp >= 0) src = xin + ((size_t)b * T_ + tp) * 64 + seg * 8;
    } else if (seg < 16) {
      dst = &lact[n * LACT_S + 64 + (seg - 8) * 8];
      src = xin + ((size_t)b * T_ + t) * 64 + (seg - 8) * 8;
    } else {
      dst = &lact[n * LACT_S + 128 + (seg - 16) * 8];
      src = cond + ((size_t)b * T_ + t) * 96 + (seg - 16) * 8;
    }
    if (src) {
#if HAVE_ASYNC_LDS
      async_cp16(src, dst);
#else
      *(uint4*)dst = *(const uint4*)src;
#endif
    } else {
      *(uint4*)dst = make_uint4(0u, 0u, 0u, 0u);
    }
  }
  WAIT_ASYNC();
  __syncthreads();

  const int lane  = tid & 31;
  const int wave  = tid >> 5;
  const int laneM = lane & 15;
  const int hiB   = lane >> 4;       // 0 / 1
  const int hi8   = hiB << 3;        // 0 / 8
  const int nB    = wave * 16 + laneM;
  const int tN    = t0 + nB;

  // ---- in_act = Wfused @ [x_prev | x | cond] + bias : M=128, K=224 ----
  v8f acc[8];
#pragma unroll
  for (int mt = 0; mt < 8; ++mt) acc[mt] = ld8f(bl + layer * 128 + mt * 16 + hi8);

#pragma unroll
  for (int kc = 0; kc < 7; ++kc) {
    const int k0 = kc * 32;
    const v16bf bf = ldB(&lact[nB * LACT_S + k0 + hiB * 16]);
#pragma unroll
    for (int mt = 0; mt < 8; ++mt) {
      const u16* ap = Wl + ((size_t)(layer * 128 + mt * 16 + laneM)) * KF + k0 + hiB * 16;
      acc[mt] = WMMA_BF16(ldA(ap), bf, acc[mt]);
    }
  }

  // ---- gated activation -> LDS [n][c] ----
#pragma unroll
  for (int mt = 0; mt < 4; ++mt) {
#pragma unroll
    for (int jp = 0; jp < 4; ++jp) {
      const float g0 = tanhf(acc[mt][2 * jp])     * (1.0f / (1.0f + __expf(-acc[mt + 4][2 * jp])));
      const float g1 = tanhf(acc[mt][2 * jp + 1]) * (1.0f / (1.0f + __expf(-acc[mt + 4][2 * jp + 1])));
      const int c = mt * 16 + hi8 + jp * 2;
      *(unsigned*)&lgat[nB * LGAT_S + c] = pack2(g0, g1);
    }
  }
  __syncthreads();

  const v16bf gA = ldB(&lgat[nB * LGAT_S + 0  + hiB * 16]);
  const v16bf gB = ldB(&lgat[nB * LGAT_S + 32 + hiB * 16]);

  // ---- residual: x_out = res_w @ acts + res_b + x ----
  if (!last) {
#pragma unroll
    for (int mt = 0; mt < 4; ++mt) {
      v8f r = ld8f(rb + layer * 64 + mt * 16 + hi8);
      const u16* a0 = Wr + ((size_t)(layer * 64 + mt * 16 + laneM)) * 64 + hiB * 16;
      r = WMMA_BF16(ldA(a0),      gA, r);
      r = WMMA_BF16(ldA(a0 + 32), gB, r);
      const size_t xoff = ((size_t)b * T_ + tN) * 64 + mt * 16 + hi8;
      uint4 xv = *(const uint4*)(xin + xoff);
      const u16* xs = (const u16*)&xv;
      uint4 ov; u16* os = (u16*)&ov;
#pragma unroll
      for (int j = 0; j < 8; ++j) os[j] = f2bf(r[j] + bf2f(xs[j]));
      *(uint4*)(xout + xoff) = ov;
    }
  }

  // ---- skip: skip_acc (+)= skip_w @ acts + skip_b ----
#pragma unroll
  for (int mt = 0; mt < 16; ++mt) {
    v8f s = ld8f(kb + layer * 256 + mt * 16 + hi8);
    const u16* a0 = Wk + ((size_t)(layer * 256 + mt * 16 + laneM)) * 64 + hiB * 16;
    s = WMMA_BF16(ldA(a0),      gA, s);
    s = WMMA_BF16(ldA(a0 + 32), gB, s);
    float* sp = skip + ((size_t)b * T_ + tN) * 256 + mt * 16 + hi8;
    if (first) {
      *(float4*)sp       = make_float4(s[0], s[1], s[2], s[3]);
      *(float4*)(sp + 4) = make_float4(s[4], s[5], s[6], s[7]);
    } else {
      float4 o0 = *(const float4*)sp, o1 = *(const float4*)(sp + 4);
      o0.x += s[0]; o0.y += s[1]; o0.z += s[2]; o0.w += s[3];
      o1.x += s[4]; o1.y += s[5]; o1.z += s[6]; o1.w += s[7];
      *(float4*)sp = o0; *(float4*)(sp + 4) = o1;
    }
  }
}

// ---------------- output head: relu -> out_w GEMM -> relu -> end_w GEMM -> shift ----------------
__global__ __launch_bounds__(128) void output_kernel(
    const float* __restrict__ skip, const u16* __restrict__ Wo,
    const u16* __restrict__ We, float* __restrict__ out)
{
  __shared__ u16 lsk[TILE_N * LOUT_S];
  __shared__ u16 lh2[TILE_N * LOUT_S];
  const int tid = threadIdx.x;
  const int b  = blockIdx.y;
  const int t0 = blockIdx.x * TILE_N;

  if (blockIdx.x == 0)
    for (int o = tid; o < 256; o += 128) out[((size_t)b * 256 + o) * T_] = 0.0f;

  for (int u = tid; u < TILE_N * 64; u += 128) {
    const int n = u >> 6, c4 = (u & 63) * 4;
    float4 v = *(const float4*)(skip + ((size_t)b * T_ + t0 + n) * 256 + c4);
    v.x = fmaxf(v.x, 0.f); v.y = fmaxf(v.y, 0.f); v.z = fmaxf(v.z, 0.f); v.w = fmaxf(v.w, 0.f);
    uint2 p; p.x = pack2(v.x, v.y); p.y = pack2(v.z, v.w);
    *(uint2*)&lsk[n * LOUT_S + c4] = p;
  }
  __syncthreads();

  const int lane  = tid & 31;
  const int wave  = tid >> 5;
  const int laneM = lane & 15;
  const int hiB   = lane >> 4;
  const int hi8   = hiB << 3;
  const int nB    = wave * 16 + laneM;

  // GEMM1: h2 = relu(out_w @ relu(skip)), M=K=256
#pragma unroll 4
  for (int mt = 0; mt < 16; ++mt) {
    v8f a = {};
#pragma unroll
    for (int kc = 0; kc < 8; ++kc) {
      const int k0 = kc * 32;
      const v16bf bf = ldB(&lsk[nB * LOUT_S + k0 + hiB * 16]);
      const u16* ap = Wo + ((size_t)(mt * 16 + laneM)) * 256 + k0 + hiB * 16;
      a = WMMA_BF16(ldA(ap), bf, a);
    }
#pragma unroll
    for (int jp = 0; jp < 4; ++jp) {
      const int c = mt * 16 + hi8 + jp * 2;
      *(unsigned*)&lh2[nB * LOUT_S + c] =
          pack2(fmaxf(a[2 * jp], 0.f), fmaxf(a[2 * jp + 1], 0.f));
    }
  }
  __syncthreads();

  // GEMM2: out = end_w @ h2, shifted right by one sample
  const int tN = t0 + nB;
#pragma unroll 4
  for (int mt = 0; mt < 16; ++mt) {
    v8f a = {};
#pragma unroll
    for (int kc = 0; kc < 8; ++kc) {
      const int k0 = kc * 32;
      const v16bf bf = ldB(&lh2[nB * LOUT_S + k0 + hiB * 16]);
      const u16* ap = We + ((size_t)(mt * 16 + laneM)) * 256 + k0 + hiB * 16;
      a = WMMA_BF16(ldA(ap), bf, a);
    }
    if (tN + 1 < T_) {
#pragma unroll
      for (int j = 0; j < 8; ++j)
        out[((size_t)b * 256 + mt * 16 + hi8 + j) * T_ + tN + 1] = a[j];
    }
  }
}

// ---------------- launch ----------------
extern "C" void kernel_launch(void* const* d_in, const int* in_sizes, int n_in,
                              void* d_out, int out_size, void* d_ws, size_t ws_size,
                              hipStream_t stream) {
  (void)in_sizes; (void)n_in; (void)out_size; (void)ws_size;
  const float* features = (const float*)d_in[0];
  const int*   fwd      = (const int*)  d_in[1];
  const float* embed    = (const float*)d_in[2];
  const float* up_w     = (const float*)d_in[3];
  const float* up_b     = (const float*)d_in[4];
  const float* cond_w   = (const float*)d_in[5];
  const float* cond_b   = (const float*)d_in[6];
  const float* dil_w    = (const float*)d_in[7];
  const float* dil_b    = (const float*)d_in[8];
  const float* res_w    = (const float*)d_in[9];
  const float* res_b    = (const float*)d_in[10];
  const float* skip_w   = (const float*)d_in[11];
  const float* skip_b   = (const float*)d_in[12];
  const float* out_w    = (const float*)d_in[13];
  const float* end_w    = (const float*)d_in[14];
  float* out = (float*)d_out;

  char* ws = (char*)d_ws;
  size_t off = 0;
  auto alloc = [&](size_t bytes) {
    char* p = ws + off;
    off = (off + bytes + 255) & ~(size_t)255;
    return p;
  };
  u16*   Wl   = (u16*)  alloc((size_t)NL_ * 128 * KF * 2);
  float* bl   = (float*)alloc((size_t)NL_ * 128 * 4);
  u16*   Wr   = (u16*)  alloc((size_t)(NL_ - 1) * 64 * 64 * 2);
  u16*   Wk   = (u16*)  alloc((size_t)NL_ * 256 * 64 * 2);
  u16*   Wo   = (u16*)  alloc((size_t)256 * 256 * 2);
  u16*   We   = (u16*)  alloc((size_t)256 * 256 * 2);
  u16*   Emb  = (u16*)  alloc((size_t)256 * 64 * 2);
  float* upT  = (float*)alloc((size_t)NC_ * KUP_ * NC_ * 4);
  u16*   cond = (u16*)  alloc((size_t)B_ * T_ * 96 * 2);
  u16*   xA   = (u16*)  alloc((size_t)B_ * T_ * 64 * 2);
  u16*   xB   = (u16*)  alloc((size_t)B_ * T_ * 64 * 2);
  float* skip = (float*)alloc((size_t)B_ * T_ * 256 * 4);

  pack_kernel<<<2048, 256, 0, stream>>>(embed, up_w, cond_w, cond_b, dil_w, dil_b,
                                        res_w, skip_w, out_w, end_w,
                                        Wl, bl, Wr, Wk, Wo, We, Emb, upT);
  upsample_kernel<<<dim3(T_ / 8, B_), 256, 0, stream>>>(features, upT, up_b, cond);
  embed_kernel<<<2048, 256, 0, stream>>>(fwd, Emb, xA);

  u16* xin = xA; u16* xout = xB;
  for (int i = 0; i < NL_; ++i) {
    layer_kernel<<<dim3(T_ / TILE_N, B_), 128, 0, stream>>>(
        Wl, bl, Wr, res_b, Wk, skip_b, cond, xin, xout, skip,
        i, 1 << (i % 8), (i == 0) ? 1 : 0, (i == NL_ - 1) ? 1 : 0);
    u16* tmp = xin; xin = xout; xout = tmp;
  }
  output_kernel<<<dim3(T_ / TILE_N, B_), 128, 0, stream>>>(skip, Wo, We, out);
}
